// Net_66829691126193
// MI455X (gfx1250) — matrix operand order repaired and checked by reading, not compiled, not run on hardware
//
#include <hip/hip_runtime.h>
#include <hip/hip_bf16.h>

typedef _Float16 half16 __attribute__((ext_vector_type(16)));
typedef float    float8 __attribute__((ext_vector_type(8)));

#define FEAT 64
#define NGRAPHS 128
#define T 256

// ---------------------------------------------------------------------------
// init: zero int degree counters + pooled-graph accumulator
// ---------------------------------------------------------------------------
__global__ void k_init(int* __restrict__ degE, float* __restrict__ g, int n_nodes) {
    int i = blockIdx.x * blockDim.x + threadIdx.x;
    if (i < n_nodes) degE[i] = 0;
    if (i < NGRAPHS * FEAT) g[i] = 0.0f;
}

// in-degree histogram (excluding self-loops)
__global__ void k_count(const int* __restrict__ dst, int* __restrict__ degE, int n_edges) {
    int e = blockIdx.x * blockDim.x + threadIdx.x;
    if (e < n_edges) atomicAdd(&degE[dst[e]], 1);
}

// block-local exclusive scan of degE -> rowstart (partial), block totals
__global__ void k_scan_block(const int* __restrict__ degE, int* __restrict__ rowstart,
                             int* __restrict__ blockSums, int n_nodes) {
    __shared__ int sbuf[T];
    int i = blockIdx.x * T + threadIdx.x;
    int v = (i < n_nodes) ? degE[i] : 0;
    sbuf[threadIdx.x] = v;
    __syncthreads();
#pragma unroll
    for (int off = 1; off < T; off <<= 1) {
        int t = (threadIdx.x >= off) ? sbuf[threadIdx.x - off] : 0;
        __syncthreads();
        sbuf[threadIdx.x] += t;
        __syncthreads();
    }
    int incl = sbuf[threadIdx.x];
    if (i < n_nodes) rowstart[i] = incl - v;            // exclusive within block
    if (threadIdx.x == T - 1) blockSums[blockIdx.x] = incl;
}

// serial exclusive scan of block totals (391 elements -> trivial)
__global__ void k_scan_sums(int* __restrict__ blockSums, int nblocks) {
    if (blockIdx.x == 0 && threadIdx.x == 0) {
        int run = 0;
        for (int b = 0; b < nblocks; ++b) { int t = blockSums[b]; blockSums[b] = run; run += t; }
    }
}

// finalize rowstart, copy to bucket cursor, dinv = rsqrt(deg+1 self loop)
__global__ void k_prep(int* __restrict__ rowstart, const int* __restrict__ blockSums,
                       int* __restrict__ cursor, const int* __restrict__ degE,
                       float* __restrict__ dinv, int n_nodes, int n_edges) {
    int i = blockIdx.x * blockDim.x + threadIdx.x;
    if (i < n_nodes) {
        int rs = rowstart[i] + blockSums[i >> 8];
        rowstart[i] = rs;
        cursor[i]   = rs;
        dinv[i]     = rsqrtf((float)degE[i] + 1.0f);
    }
    if (i == 0) rowstart[n_nodes] = n_edges;
}

// scatter edges into dst-grouped CSR buckets (int atomics only)
__global__ void k_bucket(const int* __restrict__ src, const int* __restrict__ dst,
                         int* __restrict__ cursor, int* __restrict__ esrc, int n_edges) {
    int e = blockIdx.x * blockDim.x + threadIdx.x;
    if (e < n_edges) {
        int pos = atomicAdd(&cursor[dst[e]], 1);
        esrc[pos] = src[e];
    }
}

// ---------------------------------------------------------------------------
// Y[n_rows,64] = X[n_rows,64] @ W[64,64]  via v_wmma_f32_16x16x32_f16.
// Wave32 owns a 16x64 tile; W^T staged as f16 in LDS (B frag = one 32B read).
// ---------------------------------------------------------------------------
__global__ void k_gemm64_wmma(const float* __restrict__ X, const float* __restrict__ W,
                              float* __restrict__ Y, int n_rows) {
    __shared__ _Float16 sWt[FEAT * FEAT];   // sWt[n*64+k] = (f16)W[k*64+n]
    for (int idx = threadIdx.x; idx < FEAT * FEAT; idx += blockDim.x) {
        int n = idx >> 6, k = idx & 63;
        sWt[idx] = (_Float16)W[k * FEAT + n];
    }
    __syncthreads();

    const int wave = threadIdx.x >> 5;
    const int lane = threadIdx.x & 31;
    const int row0 = (blockIdx.x * 8 + wave) * 16;
    if (row0 >= n_rows) return;

    const int m = lane & 15;                 // A-row / B,C-col within tile
    const int g = lane >> 4;                 // lane half-group

    int arow = row0 + m;
    if (arow >= n_rows) arow = n_rows - 1;
    const float4* xr = (const float4*)(X + (size_t)arow * FEAT);

    half16 a[2];
#pragma unroll
    for (int ks = 0; ks < 2; ++ks) {
        float4 f0 = xr[ks * 8 + 4 * g];
        float4 f1 = xr[ks * 8 + 4 * g + 1];
        float4 f2 = xr[ks * 8 + 4 * g + 2];
        float4 f3 = xr[ks * 8 + 4 * g + 3];
        a[ks][0]  = (_Float16)f0.x; a[ks][1]  = (_Float16)f0.y;
        a[ks][2]  = (_Float16)f0.z; a[ks][3]  = (_Float16)f0.w;
        a[ks][4]  = (_Float16)f1.x; a[ks][5]  = (_Float16)f1.y;
        a[ks][6]  = (_Float16)f1.z; a[ks][7]  = (_Float16)f1.w;
        a[ks][8]  = (_Float16)f2.x; a[ks][9]  = (_Float16)f2.y;
        a[ks][10] = (_Float16)f2.z; a[ks][11] = (_Float16)f2.w;
        a[ks][12] = (_Float16)f3.x; a[ks][13] = (_Float16)f3.y;
        a[ks][14] = (_Float16)f3.z; a[ks][15] = (_Float16)f3.w;
    }

    const bool full = (row0 + 16 <= n_rows);   // always true for this workload
#pragma unroll
    for (int nt = 0; nt < 4; ++nt) {
        float8 c;
#pragma unroll
        for (int i = 0; i < 8; ++i) c[i] = 0.0f;
#pragma unroll
        for (int ks = 0; ks < 2; ++ks) {
            const half16* bp =
                (const half16*)&sWt[(size_t)(nt * 16 + m) * FEAT + ks * 32 + g * 16];
            c = __builtin_amdgcn_wmma_f32_16x16x32_f16(
                    false, a[ks], false, *bp, (short)0, c, false, false);
        }
        // C/D layout: VGPR r -> row r + 8g, col nt*16 + m
        float* yp = Y + (size_t)(row0 + 8 * g) * FEAT + nt * 16 + m;
        if (full) {
#pragma unroll
            for (int r = 0; r < 8; ++r) yp[(size_t)r * FEAT] = c[r];
        } else {
#pragma unroll
            for (int r = 0; r < 8; ++r)
                if (row0 + 8 * g + r < n_rows) yp[(size_t)r * FEAT] = c[r];
        }
    }
}

// ---------------------------------------------------------------------------
// Fused CSR gather + self-loop + bias + ReLU. One wave per node, no atomics.
// H[d] = relu( dinv[d]*( sum_{s in N(d)} xw[s]*dinv[s] ) + xw[d]*dinv[d]^2 + b )
// ---------------------------------------------------------------------------
__global__ void k_gather(const float* __restrict__ XW, const float* __restrict__ dinv,
                         const int* __restrict__ rowstart, const int* __restrict__ esrc,
                         const float* __restrict__ bias, float* __restrict__ H,
                         int n_nodes) {
    int node = (blockIdx.x * blockDim.x + threadIdx.x) >> 5;
    int lane = threadIdx.x & 31;
    if (node >= n_nodes) return;

    int j   = rowstart[node];
    int end = rowstart[node + 1];
    float ax = 0.0f, ay = 0.0f;
    for (; j < end; ++j) {
        int s = esrc[j];                           // uniform across wave
        float ns = dinv[s];
        float2 v = *(const float2*)(XW + (size_t)s * FEAT + lane * 2);
        ax = fmaf(v.x, ns, ax);
        ay = fmaf(v.y, ns, ay);
    }
    float dn = dinv[node];
    float2 sv = *(const float2*)(XW + (size_t)node * FEAT + lane * 2);
    float2 b  = *(const float2*)(bias + lane * 2);
    ax = fmaf(ax + sv.x * dn, dn, b.x);            // (sum + xw[d]*dn)*dn + b
    ay = fmaf(ay + sv.y * dn, dn, b.y);
    float2 o;
    o.x = fmaxf(ax, 0.0f);
    o.y = fmaxf(ay, 0.0f);
    *(float2*)(H + (size_t)node * FEAT + lane * 2) = o;
}

// global_add_pool: g[batch[node], f] += h[node, f]
__global__ void k_pool(const float* __restrict__ h, const int* __restrict__ batch,
                       float* __restrict__ g, int n_nodes) {
    size_t i = (size_t)blockIdx.x * blockDim.x + threadIdx.x;
    if (i >= (size_t)n_nodes * FEAT) return;
    int node = (int)(i >> 6), f = (int)(i & 63);
    atomicAdd(&g[(size_t)batch[node] * FEAT + f], h[i]);
}

// head: out[r] = relu(g[r] @ fc1 + fc1_b) @ fc2 + fc2_b   (128 rows, 1 block)
__global__ void k_head(const float* __restrict__ g, const float* __restrict__ fc1w,
                       const float* __restrict__ fc1b, const float* __restrict__ fc2w,
                       const float* __restrict__ fc2b, float* __restrict__ out) {
    int r = threadIdx.x;
    if (r >= NGRAPHS) return;
    float row[FEAT];
#pragma unroll
    for (int k = 0; k < FEAT; ++k) row[k] = g[(size_t)r * FEAT + k];
    float o = fc2b[0];
    for (int j = 0; j < FEAT; ++j) {
        float s = fc1b[j];
#pragma unroll
        for (int k = 0; k < FEAT; ++k) s = fmaf(row[k], fc1w[k * FEAT + j], s);
        o += fmaxf(s, 0.0f) * fc2w[j];
    }
    out[r] = o;
}

// ---------------------------------------------------------------------------
extern "C" void kernel_launch(void* const* d_in, const int* in_sizes, int n_in,
                              void* d_out, int out_size, void* d_ws, size_t ws_size,
                              hipStream_t stream) {
    const float* x     = (const float*)d_in[0];
    const int*   eidx  = (const int*)d_in[1];
    const int*   batch = (const int*)d_in[2];
    const float* W1    = (const float*)d_in[3];
    const float* b1    = (const float*)d_in[4];
    const float* W2    = (const float*)d_in[5];
    const float* b2    = (const float*)d_in[6];
    const float* fc1w  = (const float*)d_in[7];
    const float* fc1b  = (const float*)d_in[8];
    const float* fc2w  = (const float*)d_in[9];
    const float* fc2b  = (const float*)d_in[10];
    float* out = (float*)d_out;

    const int n_nodes = in_sizes[0] / FEAT;     // 100000
    const int n_edges = in_sizes[1] / 2;        // 1600000
    const int* e_src = eidx;
    const int* e_dst = eidx + n_edges;

    const int nf      = n_nodes * FEAT;
    const int blkNF   = (nf + T - 1) / T;
    const int blkN    = (n_nodes + T - 1) / T;
    const int blkE    = (n_edges + T - 1) / T;
    const int blkNodeWave = (n_nodes * 32 + T - 1) / T;   // wave per node
    const int tiles   = (n_nodes + 15) / 16;
    const int blkGemm = (tiles + 7) / 8;                  // 8 waves per block

    // workspace layout
    float* A    = (float*)d_ws;                 // xw buffer    [nf]
    float* B    = A + (size_t)nf;               // h buffer     [nf]
    float* dinv = B + (size_t)nf;               // [n_nodes]
    float* g    = dinv + n_nodes;               // [128*64]
    int* degE     = (int*)(g + NGRAPHS * FEAT); // [n_nodes]
    int* rowstart = degE + n_nodes;             // [n_nodes+1]
    int* cursor   = rowstart + n_nodes + 1;     // [n_nodes]
    int* blockSum = cursor + n_nodes;           // [blkN]
    int* esrc     = blockSum + blkN;            // [n_edges]

    // --- CSR-by-destination build + symmetric norm (once, reused both layers) ---
    k_init      <<<blkN, T, 0, stream>>>(degE, g, n_nodes);
    k_count     <<<blkE, T, 0, stream>>>(e_dst, degE, n_edges);
    k_scan_block<<<blkN, T, 0, stream>>>(degE, rowstart, blockSum, n_nodes);
    k_scan_sums <<<1, 1, 0, stream>>>(blockSum, blkN);
    k_prep      <<<blkN, T, 0, stream>>>(rowstart, blockSum, cursor, degE, dinv,
                                         n_nodes, n_edges);
    k_bucket    <<<blkE, T, 0, stream>>>(e_src, e_dst, cursor, esrc, n_edges);

    // --- layer 1 ---
    k_gemm64_wmma<<<blkGemm, T, 0, stream>>>(x, W1, A, n_nodes);
    k_gather     <<<blkNodeWave, T, 0, stream>>>(A, dinv, rowstart, esrc, b1, B, n_nodes);

    // --- layer 2 ---
    k_gemm64_wmma<<<blkGemm, T, 0, stream>>>(B, W2, A, n_nodes);
    k_gather     <<<blkNodeWave, T, 0, stream>>>(A, dinv, rowstart, esrc, b2, B, n_nodes);

    // --- pool + MLP head ---
    k_pool<<<blkNF, T, 0, stream>>>(B, batch, g, n_nodes);
    k_head<<<1, NGRAPHS, 0, stream>>>(g, fc1w, fc1b, fc2w, fc2b, out);
    (void)n_in; (void)out_size; (void)ws_size;
}
